// CrossAttention_30657476559151
// MI455X (gfx1250) — matrix operand order repaired and checked by reading, not compiled, not run on hardware
//
#include <hip/hip_runtime.h>

typedef __bf16 bf16_t;
typedef __attribute__((ext_vector_type(16))) __bf16 v16bf;
typedef __attribute__((ext_vector_type(8)))  __bf16 v8bf;
typedef __attribute__((ext_vector_type(8)))  float   v8f;
typedef int v4i __attribute__((vector_size(16)));   // b128 payload type used by async-LDS builtins

#define B_  4
#define LQ_ 1024
#define LK_ 2048
#define QD_ 768
#define KD_ 512
#define H_  1024
#define NH_ 16
#define HD_ 64

// gfx1250 async global->LDS path (ASYNCcnt), guarded so the file compiles on
// toolchains where the builtins are absent (falls back to synchronous copies).
#if defined(__gfx1250__) && \
    __has_builtin(__builtin_amdgcn_global_load_async_to_lds_b128) && \
    __has_builtin(__builtin_amdgcn_s_wait_asynccnt)
#define USE_ASYNC_LDS 1
#else
#define USE_ASYNC_LDS 0
#endif

union Frag { v16bf v; v8bf h[2]; };

// A-matrix 16x32 bf16 fragment (ISA 7.12.2): lane holds row (lane%16),
// K elems [half*8, half*8+8) and [half*8+16, half*8+24). p = row base + half*8.
__device__ __forceinline__ v16bf load_a_frag(const bf16_t* p) {
    Frag f;
    f.h[0] = *(const v8bf*)(p);
    f.h[1] = *(const v8bf*)(p + 16);
    return f.v;
}
// B-matrix 32x16 bf16 fragment via B^T rows: lane holds col N=(lane%16),
// K elems [half*8, half*8+8) and [half*8+16, half*8+24). p = BT row base + half*8.
__device__ __forceinline__ v16bf load_b_frag(const bf16_t* p) {
    Frag f;
    f.h[0] = *(const v8bf*)(p);
    f.h[1] = *(const v8bf*)(p + 16);
    return f.v;
}

// Copy 16B (8 bf16) global -> LDS; async (GLOBAL_LOAD_ASYNC_TO_LDS_B128) on gfx1250.
__device__ __forceinline__ void copy16_g2l(const bf16_t* g, bf16_t* l) {
#if USE_ASYNC_LDS
    __builtin_amdgcn_global_load_async_to_lds_b128(
        (__attribute__((address_space(1))) v4i*)(const_cast<bf16_t*>(g)),
        (__attribute__((address_space(3))) v4i*)l, 0, 0);
#else
    *(v8bf*)l = *(const v8bf*)g;
#endif
}
__device__ __forceinline__ void stage_wait() {
#if USE_ASYNC_LDS
    __builtin_amdgcn_s_wait_asynccnt(0);
#endif
}

__global__ void cvt_f32_bf16(const float* __restrict__ in, bf16_t* __restrict__ out, int n) {
    int i = blockIdx.x * blockDim.x + threadIdx.x;
    int stride = gridDim.x * blockDim.x;
    for (; i < n; i += stride) out[i] = (bf16_t)in[i];
}

// in[K][N] fp32 -> out[N][K] bf16 (weight transpose so B-fragments read contiguous K)
__global__ void cvt_transpose(const float* __restrict__ in, bf16_t* __restrict__ out, int K, int N) {
    int total = K * N;
    int i = blockIdx.x * blockDim.x + threadIdx.x;
    int stride = gridDim.x * blockDim.x;
    for (; i < total; i += stride) {
        int n = i / K, k = i - n * K;
        out[i] = (bf16_t)in[k * N + n];
    }
}

// C = A[M][K] @ B[K][N] + bias, with B given transposed BT[N][K] (bf16).
// Double-buffered LDS pipeline: stage tile kt+1 (async if available) while
// running WMMAs on tile kt; single barrier per K-step.
// MODE 0: write bf16 head-split layout [b][NH][L][HD]; MODE 1: plain f32 [M][N].
#define LDA 40   // padded LDS strides (bank-conflict relief, keep 16B alignment)
template<int MODE>
__global__ void __launch_bounds__(256)
gemm_wmma(const bf16_t* __restrict__ A, const bf16_t* __restrict__ BT,
          const float* __restrict__ bias, void* __restrict__ outp,
          int M, int N, int K, int L) {
    __shared__ bf16_t lsA[2][128 * LDA];
    __shared__ bf16_t lsB[2][64 * LDA];

    const int tid   = threadIdx.x;
    const int lane  = tid & 31;
    const int wave  = tid >> 5;          // 8 waves
    const int r     = lane & 15;
    const int half  = lane >> 4;
    const int waveM = wave & 3;          // 4 waves along M
    const int waveN = wave >> 2;         // 2 waves along N

    const int mBlk = blockIdx.y * 128;
    const int nBlk = blockIdx.x * 64;

    v8f c[2][2] = {};

    const int aRow = tid >> 1;           // 0..127
    const int aCol = (tid & 1) * 16;     // 0 / 16
    const int bRow = tid >> 2;           // 0..63
    const int bCol = (tid & 3) * 8;      // 0,8,16,24

    const bf16_t* gaBase = A  + (size_t)(mBlk + aRow) * K + aCol;
    const bf16_t* gbBase = BT + (size_t)(nBlk + bRow) * K + bCol;

    const int kTiles = K >> 5;

    // prologue: stage tile 0 into buffer 0
    copy16_g2l(gaBase,     &lsA[0][aRow * LDA + aCol]);
    copy16_g2l(gaBase + 8, &lsA[0][aRow * LDA + aCol + 8]);
    copy16_g2l(gbBase,     &lsB[0][bRow * LDA + bCol]);
    stage_wait();
    __syncthreads();

    int buf = 0;
    for (int kt = 0; kt < kTiles; ++kt) {
        if (kt + 1 < kTiles) {           // stage next tile into the other buffer
            const int k0 = (kt + 1) << 5;
            const bf16_t* ga = gaBase + k0;
            const bf16_t* gb = gbBase + k0;
            copy16_g2l(ga,     &lsA[buf ^ 1][aRow * LDA + aCol]);
            copy16_g2l(ga + 8, &lsA[buf ^ 1][aRow * LDA + aCol + 8]);
            copy16_g2l(gb,     &lsB[buf ^ 1][bRow * LDA + bCol]);
            if (kt + 2 < kTiles) {       // global_prefetch_b8 two tiles ahead
                __builtin_prefetch(ga + 32, 0, 1);
                __builtin_prefetch(gb + 32, 0, 1);
            }
        }

        v16bf a[2], b[2];
#pragma unroll
        for (int mi = 0; mi < 2; ++mi)
            a[mi] = load_a_frag(&lsA[buf][(waveM * 32 + mi * 16 + r) * LDA + half * 8]);
#pragma unroll
        for (int ni = 0; ni < 2; ++ni)
            b[ni] = load_b_frag(&lsB[buf][(waveN * 32 + ni * 16 + r) * LDA + half * 8]);
#pragma unroll
        for (int mi = 0; mi < 2; ++mi)
#pragma unroll
            for (int ni = 0; ni < 2; ++ni)
                c[mi][ni] = __builtin_amdgcn_wmma_f32_16x16x32_bf16(
                    false, a[mi], false, b[ni], (short)0, c[mi][ni], false, false);

        stage_wait();                    // next-tile stage complete
        __syncthreads();                 // all waves done reading buf
        buf ^= 1;
    }

#pragma unroll
    for (int mi = 0; mi < 2; ++mi) {
#pragma unroll
        for (int ni = 0; ni < 2; ++ni) {
            const int nCol = nBlk + waveN * 32 + ni * 16 + r;
            const float bv = bias[nCol];
#pragma unroll
            for (int i = 0; i < 8; ++i) {
                const int mRow = mBlk + waveM * 32 + mi * 16 + half * 8 + i;
                const float val = c[mi][ni][i] + bv;
                if (MODE == 0) {
                    const int bb = mRow / L;
                    const int l  = mRow - bb * L;
                    const int hh = nCol >> 6;       // HD=64
                    const int d  = nCol & 63;
                    ((bf16_t*)outp)[(((size_t)(bb * NH_ + hh) * L) + l) * HD_ + d] = (bf16_t)val;
                } else {
                    ((float*)outp)[(size_t)mRow * N + nCol] = val;
                }
            }
        }
    }
}

// Flash attention: grid = B*NH*(LQ/64), block = 128 (4 waves, 16 q-rows each).
#define LDT 72   // padded LDS stride for 64-wide tiles
__global__ void __launch_bounds__(128)
attn_wmma(const bf16_t* __restrict__ qh, const bf16_t* __restrict__ kh,
          const bf16_t* __restrict__ vh, bf16_t* __restrict__ att) {
    __shared__ bf16_t lsK [64 * LDT];       // K chunk  [key][d]   (== B^T for S)
    __shared__ bf16_t lsVT[64 * LDT];       // V chunk  [d][key]   (== B^T for P@V)
    __shared__ bf16_t lsP [4][16 * LDT];    // per-wave P tile [qrow][key]

    const int tid  = threadIdx.x;
    const int lane = tid & 31;
    const int wave = tid >> 5;
    const int r    = lane & 15;
    const int half = lane >> 4;

    const int qTiles = LQ_ / 64;
    const int bh = blockIdx.x / qTiles;
    const int qt = blockIdx.x - bh * qTiles;
    const int b  = bh / NH_;
    const int h  = bh - b * NH_;

    const bf16_t* qb = qh + (size_t)bh * LQ_ * HD_;
    const bf16_t* kb = kh + (size_t)bh * LK_ * HD_;
    const bf16_t* vb = vh + (size_t)bh * LK_ * HD_;

    const int q0 = qt * 64 + wave * 16;

    // q fragments for the two 32-wide K-steps over HD=64, scale 1/sqrt(64) folded in
    v16bf aQ[2];
#pragma unroll
    for (int s = 0; s < 2; ++s) {
        const bf16_t* p = qb + (size_t)(q0 + r) * HD_ + s * 32 + half * 8;
        Frag f;
        f.h[0] = *(const v8bf*)(p);
        f.h[1] = *(const v8bf*)(p + 16);
#pragma unroll
        for (int e = 0; e < 16; ++e)
            f.v[e] = (bf16_t)((float)f.v[e] * 0.125f);
        aQ[s] = f.v;
    }

    float mrow[8], lrow[8];
    v8f o[4] = {};
#pragma unroll
    for (int i = 0; i < 8; ++i) { mrow[i] = -3.0e30f; lrow[i] = 0.f; }

    for (int kc = 0; kc < LK_ / 64; ++kc) {
        // stage K [64][64] row-major and V transposed [d][key]
        const int row = tid >> 1;            // 0..63
        const int d0  = (tid & 1) * 32;
        const bf16_t* gk = kb + (size_t)(kc * 64 + row) * HD_ + d0;
#pragma unroll
        for (int u = 0; u < 4; ++u)
            *(v8bf*)&lsK[row * LDT + d0 + u * 8] = *(const v8bf*)(gk + u * 8);
        const bf16_t* gv = vb + (size_t)(kc * 64 + row) * HD_ + d0;
        v8bf tv[4];
#pragma unroll
        for (int u = 0; u < 4; ++u) tv[u] = *(const v8bf*)(gv + u * 8);
#pragma unroll
        for (int u = 0; u < 4; ++u)
#pragma unroll
            for (int e = 0; e < 8; ++e)
                lsVT[(d0 + u * 8 + e) * LDT + row] = tv[u][e];
        __syncthreads();

        // S = q @ k^T  (4 key-tiles of 16, 2 K-steps over HD)
        v8f sAcc[4];
#pragma unroll
        for (int j = 0; j < 4; ++j) {
            sAcc[j] = (v8f){};
#pragma unroll
            for (int s = 0; s < 2; ++s) {
                v16bf bK = load_b_frag(&lsK[(j * 16 + r) * LDT + s * 32 + half * 8]);
                sAcc[j] = __builtin_amdgcn_wmma_f32_16x16x32_bf16(
                    false, aQ[s], false, bK, (short)0, sAcc[j], false, false);
            }
        }

        // online softmax: row m = half*8+i lives in one 16-lane half (C layout)
#pragma unroll
        for (int i = 0; i < 8; ++i) {
            float v0 = sAcc[0][i], v1 = sAcc[1][i], v2 = sAcc[2][i], v3 = sAcc[3][i];
            float mx = fmaxf(fmaxf(v0, v1), fmaxf(v2, v3));
#pragma unroll
            for (int off = 1; off < 16; off <<= 1)
                mx = fmaxf(mx, __shfl_xor(mx, off, 32));
            const float mnew  = fmaxf(mrow[i], mx);
            const float alpha = __expf(mrow[i] - mnew);
            const float p0 = __expf(v0 - mnew), p1 = __expf(v1 - mnew);
            const float p2 = __expf(v2 - mnew), p3 = __expf(v3 - mnew);
            float ls = p0 + p1 + p2 + p3;
#pragma unroll
            for (int off = 1; off < 16; off <<= 1)
                ls += __shfl_xor(ls, off, 32);
            lrow[i] = lrow[i] * alpha + ls;
            mrow[i] = mnew;
#pragma unroll
            for (int t = 0; t < 4; ++t) o[t][i] *= alpha;
            const int prow = half * 8 + i;
            lsP[wave][prow * LDT +  0 + r] = (bf16_t)p0;
            lsP[wave][prow * LDT + 16 + r] = (bf16_t)p1;
            lsP[wave][prow * LDT + 32 + r] = (bf16_t)p2;
            lsP[wave][prow * LDT + 48 + r] = (bf16_t)p3;
        }

        // O += P @ V  (same-wave LDS write->read is in-order per ISA)
#pragma unroll
        for (int s = 0; s < 2; ++s) {
            v16bf aP = load_a_frag(&lsP[wave][r * LDT + s * 32 + half * 8]);
#pragma unroll
            for (int t = 0; t < 4; ++t) {
                v16bf bV = load_b_frag(&lsVT[(t * 16 + r) * LDT + s * 32 + half * 8]);
                o[t] = __builtin_amdgcn_wmma_f32_16x16x32_bf16(
                    false, aP, false, bV, (short)0, o[t], false, false);
            }
        }
        __syncthreads();
    }

    // normalize and write att in [B][LQ][H] interleaved-head layout (bf16)
#pragma unroll
    for (int t = 0; t < 4; ++t) {
#pragma unroll
        for (int i = 0; i < 8; ++i) {
            const int lq = q0 + half * 8 + i;
            const float val = o[t][i] / lrow[i];
            att[((size_t)(b * LQ_) + lq) * H_ + h * HD_ + t * 16 + r] = (bf16_t)val;
        }
    }
}

extern "C" void kernel_launch(void* const* d_in, const int* in_sizes, int n_in,
                              void* d_out, int out_size, void* d_ws, size_t ws_size,
                              hipStream_t stream) {
    (void)in_sizes; (void)n_in; (void)out_size; (void)ws_size;
    const float* query = (const float*)d_in[0];
    const float* key   = (const float*)d_in[1];
    const float* value = (const float*)d_in[2];
    const float* Wq = (const float*)d_in[3];
    const float* bq = (const float*)d_in[4];
    const float* Wk = (const float*)d_in[5];
    const float* bk = (const float*)d_in[6];
    const float* Wv = (const float*)d_in[7];
    const float* bv = (const float*)d_in[8];
    const float* Wo = (const float*)d_in[9];
    const float* bo = (const float*)d_in[10];
    float* out = (float*)d_out;

    char* w = (char*)d_ws;
    auto carve = [&](size_t elems) {
        bf16_t* p = (bf16_t*)w;
        w += (elems * sizeof(bf16_t) + 255) & ~(size_t)255;
        return p;
    };
    bf16_t* qbf = carve((size_t)B_ * LQ_ * QD_);
    bf16_t* kbf = carve((size_t)B_ * LK_ * KD_);
    bf16_t* vbf = carve((size_t)B_ * LK_ * KD_);
    bf16_t* WqT = carve((size_t)QD_ * H_);
    bf16_t* WkT = carve((size_t)KD_ * H_);
    bf16_t* WvT = carve((size_t)KD_ * H_);
    bf16_t* WoT = carve((size_t)H_ * H_);
    bf16_t* qhd = carve((size_t)B_ * LQ_ * H_);
    bf16_t* khd = carve((size_t)B_ * LK_ * H_);
    bf16_t* vhd = carve((size_t)B_ * LK_ * H_);
    bf16_t* att = carve((size_t)B_ * LQ_ * H_);

    cvt_f32_bf16<<<2048, 256, 0, stream>>>(query, qbf, B_ * LQ_ * QD_);
    cvt_f32_bf16<<<2048, 256, 0, stream>>>(key,   kbf, B_ * LK_ * KD_);
    cvt_f32_bf16<<<2048, 256, 0, stream>>>(value, vbf, B_ * LK_ * KD_);
    cvt_transpose<<<1024, 256, 0, stream>>>(Wq, WqT, QD_, H_);
    cvt_transpose<<<1024, 256, 0, stream>>>(Wk, WkT, KD_, H_);
    cvt_transpose<<<1024, 256, 0, stream>>>(Wv, WvT, KD_, H_);
    cvt_transpose<<<1024, 256, 0, stream>>>(Wo, WoT, H_, H_);

    gemm_wmma<0><<<dim3(H_ / 64, (B_ * LQ_) / 128), 256, 0, stream>>>(
        qbf, WqT, bq, qhd, B_ * LQ_, H_, QD_, LQ_);
    gemm_wmma<0><<<dim3(H_ / 64, (B_ * LK_) / 128), 256, 0, stream>>>(
        kbf, WkT, bk, khd, B_ * LK_, H_, KD_, LK_);
    gemm_wmma<0><<<dim3(H_ / 64, (B_ * LK_) / 128), 256, 0, stream>>>(
        vbf, WvT, bv, vhd, B_ * LK_, H_, KD_, LK_);

    attn_wmma<<<B_ * NH_ * (LQ_ / 64), 128, 0, stream>>>(qhd, khd, vhd, att);

    gemm_wmma<1><<<dim3(H_ / 64, (B_ * LQ_) / 128), 256, 0, stream>>>(
        att, WoT, bo, out, B_ * LQ_, H_, H_, LQ_);
}
